// Screen_59493886984836
// MI455X (gfx1250) — 8-bit matrix operand order repaired and checked by reading, compiled, not run on hardware
//
#include <hip/hip_runtime.h>

typedef __attribute__((ext_vector_type(16))) int   v16i;
typedef __attribute__((ext_vector_type(8)))  float v8f;

#define SCR_W   1280
#define SCR_H   720
#define TILE_L  16
#define NBW     80
#define NBH     45
#define NTILES  (NBW * NBH)
#define NPTS    32768
#define KDIM    128
#define NBLK_N  (NPTS / 16)
#define NBLK_T  (NTILES / 16)
#define FP8_ONE 0x38u

__global__ void __launch_bounds__(256) build_B_kernel(
    const float* __restrict__ pos2d, const float* __restrict__ radius,
    unsigned int* __restrict__ bbuf) {
  int i    = blockIdx.x * blockDim.x + threadIdx.x;
  int nblk = i >> 5;
  int lane = i & 31;
  int nn   = lane & 15;
  int hi   = lane >> 4;
  int n    = nblk * 16 + nn;

  float px = pos2d[2 * n + 0];
  float py = pos2d[2 * n + 1];
  float r  = radius[n];
  int xmin = (int)fminf(fmaxf(px - r, 0.0f), (float)SCR_W);
  int xmax = (int)fminf(fmaxf(px + r, 0.0f), (float)SCR_W);
  int ymin = (int)fminf(fmaxf(py - r, 0.0f), (float)SCR_H);
  int ymax = (int)fminf(fmaxf(py + r, 0.0f), (float)SCR_H);

  unsigned int* dst = bbuf + (size_t)i * 16;
#pragma unroll
  for (int v = 0; v < 16; ++v) {
    unsigned int w = 0;
#pragma unroll
    for (int b = 0; b < 4; ++b) {
      int k = (v >> 2) * 32 + hi * 16 + (v & 3) * 4 + b;
      int bit = 0;
      if (k < NBW) {
        int a = max(xmin, k * TILE_L);
        int c = min(xmax, k * TILE_L + TILE_L);
        bit = (c > a);
      } else if (k < NBW + NBH) {
        int yi = k - NBW;
        int a = max(ymin, yi * TILE_L);
        int c = min(ymax, yi * TILE_L + TILE_L);
        bit = (c > a);
      }
      if (bit) w |= FP8_ONE << (8 * b);
    }
    dst[v] = w;
  }
}

__global__ void __launch_bounds__(256) wmma_expand_kernel(
    const v16i* __restrict__ bbuf, float* __restrict__ out) {
  int lane = threadIdx.x & 31;
  int wave = threadIdx.x >> 5;
  int m    = lane & 15;
  int hi   = lane >> 4;
  int tblk = blockIdx.y;

  int t  = tblk * 16 + m;
  int kx = t / NBH;
  int ky = NBW + (t - (t / NBH) * NBH);

  v16i A;
#pragma unroll
  for (int v = 0; v < 16; ++v) {
    int vp = v & 7;
    int kstart = (v >> 2) * 32 + hi * 16 + (v & 3) * 4;
    unsigned int w = 0;
#pragma unroll
    for (int b = 0; b < 4; ++b) {
      int k = kstart + b;
      if (k == kx || k == ky) w |= FP8_ONE << (8 * b);
    }
    A[v] = (int)w;
  }

#pragma unroll
  for (int it = 0; it < 8; ++it) {
    int nblk = blockIdx.x * 64 + wave * 8 + it;
    v16i B = bbuf[(size_t)nblk * 32 + lane];
    v8f  C = {};
    v8f  D = __builtin_amdgcn_wmma_f32_16x16x128_fp8_fp8(
        A, B, (short)0, C, false, false);

    int    n    = nblk * 16 + m;
    size_t base = (size_t)(tblk * 16 + hi * 8) * NPTS + (size_t)n;
#pragma unroll
    for (int r = 0; r < 8; ++r) {
      out[base + (size_t)r * NPTS] = (D[r] > 1.5f) ? 1.0f : 0.0f;
    }
  }
}

__global__ void __launch_bounds__(256) fallback_kernel(
    const float* __restrict__ pos2d, const float* __restrict__ radius,
    float* __restrict__ out) {
  size_t idx = (size_t)blockIdx.x * blockDim.x + threadIdx.x;
  if (idx >= (size_t)NTILES * NPTS) return;
  int t = (int)(idx / NPTS);
  int n = (int)(idx - (size_t)t * NPTS);
  int xi = t / NBH, yi = t - xi * NBH;
  float px = pos2d[2 * n], py = pos2d[2 * n + 1], r = radius[n];
  int xmin = (int)fminf(fmaxf(px - r, 0.0f), (float)SCR_W);
  int xmax = (int)fminf(fmaxf(px + r, 0.0f), (float)SCR_W);
  int ymin = (int)fminf(fmaxf(py - r, 0.0f), (float)SCR_H);
  int ymax = (int)fminf(fmaxf(py + r, 0.0f), (float)SCR_H);
  int ax = max(xmin, xi * TILE_L), cx = min(xmax, xi * TILE_L + TILE_L);
  int ay = max(ymin, yi * TILE_L), cy = min(ymax, yi * TILE_L + TILE_L);
  out[idx] = ((cx > ax) && (cy > ay)) ? 1.0f : 0.0f;
}

extern "C" void kernel_launch(void* const* d_in, const int* in_sizes, int n_in,
                              void* d_out, int out_size, void* d_ws,
                              size_t ws_size, hipStream_t stream) {
  const float* pos2d  = (const float*)d_in[0];
  const float* radius = (const float*)d_in[1];
  float*       out    = (float*)d_out;

  const size_t bbuf_bytes = (size_t)NBLK_N * 32 * 64;
  if (ws_size >= bbuf_bytes) {
    unsigned int* bbuf = (unsigned int*)d_ws;
    build_B_kernel<<<dim3((NBLK_N * 32) / 256), dim3(256), 0, stream>>>(
        pos2d, radius, bbuf);
    wmma_expand_kernel<<<dim3(32, NBLK_T), dim3(256), 0, stream>>>(
        (const v16i*)d_ws, out);
  } else {
    size_t total  = (size_t)NTILES * NPTS;
    size_t blocks = (total + 255) / 256;
    fallback_kernel<<<dim3((unsigned)blocks), dim3(256), 0, stream>>>(
        pos2d, radius, out);
  }
}
